// Model_32933809225895
// MI455X (gfx1250) — compile-verified, matrix operand order checked by reading, not run on hardware
//
#include <hip/hip_runtime.h>

// ---------------------------------------------------------------------------
// CDNA5 (gfx1250) sparse-UNet forward.
// Core op: gather-GEMM-scatter using v_wmma_f32_16x16x32_bf16 (wave32).
// A tile: 16 gathered feature rows x 32 K (bf16), rows clamped (not masked)
// so the hot loop has no EXEC manipulation.  Each wave computes TWO 16-col
// output tiles per A load (2 WMMAs per gathered fragment).  B tiles come
// from a pre-transposed bf16 weight [Cout][Kpad].  Accumulate f32, scatter
// race-free (per-offset rulebooks have unique outputs; offsets serialize on
// the stream), so no atomics are needed.
// ---------------------------------------------------------------------------

typedef __attribute__((ext_vector_type(16))) __bf16 v16bf;
typedef __attribute__((ext_vector_type(8)))  float  v8f;

#define BN_EPS 1e-4f

__device__ __forceinline__ unsigned short f2bf(float f) {
  unsigned u = __float_as_uint(f);
  unsigned r = ((u >> 16) & 1u) + 0x7FFFu;      // round-to-nearest-even
  return (unsigned short)((u + r) >> 16);
}
__device__ __forceinline__ float bf2f(unsigned short s) {
  return __uint_as_float(((unsigned)s) << 16);
}

union ABFrag { v16bf v; uint4 q[2]; };

// One wave per block.  blockIdx.x: 16-row pair tile, blockIdx.y: 32-col group.
// 16-bit A layout (ISA 7.12.2): lane<16 holds row=lane, K = kk+{0..7} in
// elems 0..7 and kk+{16..23} in elems 8..15; lanes 16..31 same rows with
// base 8 -> two contiguous 16-byte loads per lane per 32-K chunk.  B uses
// the symmetric layout with "row" = output column (weights pre-transposed).
__global__ void __launch_bounds__(32)
gather_gemm_scatter_wmma(const unsigned short* __restrict__ xbf, // [n_in][kpad] bf16
                         const unsigned short* __restrict__ wt,  // [cout][kpad] bf16
                         const int* __restrict__ in_idx,
                         const int* __restrict__ out_idx,
                         int n_pairs, int kpad, int cout,
                         float* __restrict__ out, int out_stride, int col_off)
{
  const int lane = threadIdx.x;
  const int half = lane >> 4;
  const int l16  = lane & 15;
  // Row clamp instead of masking: D row m depends only on A row m, and we
  // never scatter rows with pair index >= n_pairs, so clamped rows are inert.
  int prow = blockIdx.x * 16 + l16;
  if (prow >= n_pairs) prow = n_pairs - 1;
  const int arow  = in_idx[prow];
  const int col0  = blockIdx.y * 32 + l16;             // first col tile
  const bool has1 = (blockIdx.y * 32 + 32) <= cout;    // uniform per block
  const int colB1 = has1 ? (col0 + 16) : col0;         // clamp B ptr if absent
  const int kbase = half * 8;

  const unsigned short* ap0 = xbf + (size_t)arow * kpad + kbase;
  const unsigned short* bp0 = wt + (size_t)col0  * kpad + kbase;
  const unsigned short* bp1 = wt + (size_t)colB1 * kpad + kbase;

  v8f acc0 = {}, acc1 = {};
  for (int kk = 0; kk < kpad; kk += 32) {
    ABFrag A, B0, B1;
    const uint4* ap = (const uint4*)(ap0 + kk);
    A.q[0] = ap[0];                 // K = kk+kbase   .. +7
    A.q[1] = ap[2];                 // K = kk+kbase+16.. +23
    const uint4* bq0 = (const uint4*)(bp0 + kk);
    B0.q[0] = bq0[0];
    B0.q[1] = bq0[2];
    const uint4* bq1 = (const uint4*)(bp1 + kk);
    B1.q[0] = bq1[0];
    B1.q[1] = bq1[2];
    __builtin_prefetch(ap0 + kk + 32, 0, 1);    // global_prefetch_b8
    acc0 = __builtin_amdgcn_wmma_f32_16x16x32_bf16(false, A.v, false, B0.v,
                                                   (short)0, acc0, false, false);
    acc1 = __builtin_amdgcn_wmma_f32_16x16x32_bf16(false, A.v, false, B1.v,
                                                   (short)0, acc1, false, false);
  }

  // D layout: lane holds one column; element v holds row m = 8*half + v.
  const int tbase = blockIdx.x * 16 + half * 8;
#pragma unroll
  for (int v = 0; v < 8; ++v) {
    int p = tbase + v;
    if (p < n_pairs) {
      int o = out_idx[p];
      float* dst = out + (size_t)o * out_stride + col_off + col0;
      dst[0] += acc0[v];                 // unique (o,col) within this launch
      if (has1) dst[16] += acc1[v];
    }
  }
}

// Per-channel BN statistics -> fused (scale, shift).  One block per channel.
__global__ void __launch_bounds__(256)
bn_stats(const float* __restrict__ x, int n, int c,
         const float* __restrict__ g, const float* __restrict__ b,
         float* __restrict__ scale, float* __restrict__ shift)
{
  __shared__ float sA[256], sB[256];
  const int ch = blockIdx.x;
  float s = 0.f, s2 = 0.f;
  for (int i = threadIdx.x; i < n; i += 256) {
    float v = x[(size_t)i * c + ch];
    s += v; s2 += v * v;
  }
  sA[threadIdx.x] = s; sB[threadIdx.x] = s2;
  __syncthreads();
  for (int o = 128; o > 0; o >>= 1) {
    if (threadIdx.x < o) {
      sA[threadIdx.x] += sA[threadIdx.x + o];
      sB[threadIdx.x] += sB[threadIdx.x + o];
    }
    __syncthreads();
  }
  if (threadIdx.x == 0) {
    float inv = 1.f / (float)n;
    float mu  = sA[0] * inv;
    float var = sB[0] * inv - mu * mu;
    float sc  = g[ch] * rsqrtf(var + BN_EPS);
    scale[ch] = sc;
    shift[ch] = b[ch] - mu * sc;
  }
}

// y = relu(x*scale+shift), written as bf16 with K padded to kpad (zeros).
__global__ void __launch_bounds__(256)
bn_apply(const float* __restrict__ x, int n, int c, int kpad,
         const float* __restrict__ scale, const float* __restrict__ shift,
         unsigned short* __restrict__ ybf)
{
  size_t total  = (size_t)n * kpad;
  size_t stride = (size_t)gridDim.x * blockDim.x;
  for (size_t i = (size_t)blockIdx.x * blockDim.x + threadIdx.x; i < total; i += stride) {
    int    ch  = (int)(i % kpad);
    size_t row = i / kpad;
    float v = 0.f;
    if (ch < c) {
      v = x[row * c + ch] * scale[ch] + shift[ch];
      v = v > 0.f ? v : 0.f;
    }
    ybf[i] = f2bf(v);
  }
}

// W [KO][cin][cout] f32 -> transposed padded bf16 [KO][cout][kpad].
__global__ void __launch_bounds__(256)
wconv_bf16t(const float* __restrict__ w, int KO, int cin, int cout, int kpad,
            unsigned short* __restrict__ wt)
{
  size_t total  = (size_t)KO * cout * kpad;
  size_t stride = (size_t)gridDim.x * blockDim.x;
  for (size_t i = (size_t)blockIdx.x * blockDim.x + threadIdx.x; i < total; i += stride) {
    int    kk = (int)(i % kpad);
    size_t t  = i / kpad;
    int    co = (int)(t % cout);
    int    ko = (int)(t / cout);
    float v = (kk < cin) ? w[((size_t)ko * cin + kk) * cout + co] : 0.f;
    wt[i] = f2bf(v);
  }
}

// features (n x 3) f32 -> bf16 padded to 32 channels.
__global__ void __launch_bounds__(256)
feat2bf(const float* __restrict__ f, int n, unsigned short* __restrict__ o)
{
  size_t total  = (size_t)n * 32;
  size_t stride = (size_t)gridDim.x * blockDim.x;
  for (size_t i = (size_t)blockIdx.x * blockDim.x + threadIdx.x; i < total; i += stride) {
    int    ch  = (int)(i & 31);
    size_t row = i >> 5;
    o[i] = (ch < 3) ? f2bf(f[row * 3 + ch]) : (unsigned short)0;
  }
}

// dst[row][coff+ch] = src[row][ch]  (fill skip-connection half of concat)
__global__ void __launch_bounds__(256)
copy_cols(const float* __restrict__ src, int n, int c,
          float* __restrict__ dst, int dstride, int coff)
{
  size_t total  = (size_t)n * c;
  size_t stride = (size_t)gridDim.x * blockDim.x;
  for (size_t i = (size_t)blockIdx.x * blockDim.x + threadIdx.x; i < total; i += stride) {
    int    ch  = (int)(i % c);
    size_t row = i / c;
    dst[row * dstride + coff + ch] = src[i];
  }
}

// final 16 -> 3 linear (too skinny for WMMA; VALU is fine, it's ~29 MFLOP).
__global__ void __launch_bounds__(256)
linear_out(const unsigned short* __restrict__ xbf, int n,
           const float* __restrict__ lw, const float* __restrict__ lb,
           float* __restrict__ out)
{
  int i = blockIdx.x * blockDim.x + threadIdx.x;
  if (i >= n) return;
  float a0 = lb[0], a1 = lb[1], a2 = lb[2];
#pragma unroll
  for (int k = 0; k < 16; ++k) {
    float v = bf2f(xbf[(size_t)i * 32 + k]);
    a0 += v * lw[k * 3 + 0];
    a1 += v * lw[k * 3 + 1];
    a2 += v * lw[k * 3 + 2];
  }
  out[(size_t)i * 3 + 0] = a0;
  out[(size_t)i * 3 + 1] = a1;
  out[(size_t)i * 3 + 2] = a2;
}

// ---------------------------------------------------------------------------
// Host orchestration.
// Input flattening assumption (insertion order of setup_inputs() pytrees):
//   coords, features,
//   rules.sm[0..6]  : 27 "in" arrays then 27 "out" arrays per level,
//   rules.down[0..5]: 8 "idx" arrays then 8 "parent" arrays per level,
//   params.w0,
//   params.levels[0..5]: pre_g,pre_b,pre_w,down_g,down_b,down_w,
//                        up_g,up_b,up_w,post_g,post_b,post_w,
//   params.levels[6]   : pre_g,pre_b,pre_w,
//   fin_g, fin_b, lin_w, lin_b                         => 556 leaves total.
// ---------------------------------------------------------------------------

extern "C" void kernel_launch(void* const* d_in, const int* in_sizes, int n_in,
                              void* d_out, int out_size, void* d_ws, size_t ws_size,
                              hipStream_t stream)
{
  (void)out_size; (void)ws_size;
  if (n_in < 556) return;

  const int NLEV = 7;
  int cur = 0;

  cur++;                                                   // coords (unused)
  const float* features = (const float*)d_in[cur];
  const int N0 = in_sizes[cur] / 3;
  cur++;

  const int* sm_in[7][27]; const int* sm_out[7][27]; int sm_n[7][27];
  for (int l = 0; l < NLEV; ++l) {
    for (int k = 0; k < 27; ++k) { sm_in[l][k]  = (const int*)d_in[cur]; sm_n[l][k] = in_sizes[cur]; cur++; }
    for (int k = 0; k < 27; ++k) { sm_out[l][k] = (const int*)d_in[cur]; cur++; }
  }
  const int* dn_idx[6][8]; const int* dn_par[6][8]; int dn_n[6][8];
  for (int l = 0; l < 6; ++l) {
    for (int k = 0; k < 8; ++k) { dn_idx[l][k] = (const int*)d_in[cur]; dn_n[l][k] = in_sizes[cur]; cur++; }
    for (int k = 0; k < 8; ++k) { dn_par[l][k] = (const int*)d_in[cur]; cur++; }
  }
  const float* w0 = (const float*)d_in[cur++];
  struct Lv {
    const float *pre_g, *pre_b, *pre_w;
    const float *down_g, *down_b, *down_w;
    const float *up_g, *up_b, *up_w;
    const float *post_g, *post_b, *post_w;
  } lv[7] = {};
  for (int l = 0; l < NLEV; ++l) {
    lv[l].pre_g = (const float*)d_in[cur++];
    lv[l].pre_b = (const float*)d_in[cur++];
    lv[l].pre_w = (const float*)d_in[cur++];
    if (l < NLEV - 1) {
      lv[l].down_g = (const float*)d_in[cur++];
      lv[l].down_b = (const float*)d_in[cur++];
      lv[l].down_w = (const float*)d_in[cur++];
      lv[l].up_g   = (const float*)d_in[cur++];
      lv[l].up_b   = (const float*)d_in[cur++];
      lv[l].up_w   = (const float*)d_in[cur++];
      lv[l].post_g = (const float*)d_in[cur++];
      lv[l].post_b = (const float*)d_in[cur++];
      lv[l].post_w = (const float*)d_in[cur++];
    }
  }
  const float* fin_g = (const float*)d_in[cur++];
  const float* fin_b = (const float*)d_in[cur++];
  const float* lin_w = (const float*)d_in[cur++];
  const float* lin_b = (const float*)d_in[cur++];

  int Ns[7], C[7];
  for (int l = 0; l < NLEV; ++l) { Ns[l] = sm_n[l][13]; C[l] = 16 * (l + 1); }
  auto kp = [](int c) { return (c + 31) & ~31; };

  // ---- bump allocator over workspace ----
  char* wbase = (char*)d_ws;
  size_t woff = 0;
  auto alloc = [&](size_t bytes) -> void* {
    void* p = wbase + woff;
    woff = (woff + bytes + 255) & ~(size_t)255;
    return p;
  };

  float* ss = (float*)alloc(512 * sizeof(float));            // scale | shift
  unsigned short* in0 = (unsigned short*)alloc((size_t)N0 * 32 * 2);
  float* x0 = (float*)alloc((size_t)N0 * 16 * sizeof(float));
  unsigned short* buf[7];
  float* xA[7];
  size_t maxXd = 0, maxCat = 0, maxXp = 0;
  for (int l = 0; l < NLEV; ++l) {
    buf[l] = (unsigned short*)alloc((size_t)Ns[l] * kp(2 * C[l]) * 2);
    xA[l]  = (float*)alloc((size_t)Ns[l] * C[l] * sizeof(float));
    if (l > 0 && (size_t)Ns[l] * C[l] > maxXd) maxXd = (size_t)Ns[l] * C[l];
    size_t cat = (size_t)Ns[l] * 2 * C[l];
    if (cat > maxCat) maxCat = cat;
    if ((size_t)Ns[l] * C[l] > maxXp) maxXp = (size_t)Ns[l] * C[l];
  }
  float* xd  = (float*)alloc(maxXd  * sizeof(float));
  float* cat = (float*)alloc(maxCat * sizeof(float));
  float* xP  = (float*)alloc(maxXp  * sizeof(float));

  // ---- launch helpers ----
  auto gblocks = [](size_t total) {
    size_t b = (total + 255) / 256;
    if (b > 4096) b = 4096;
    if (b < 1) b = 1;
    return (unsigned)b;
  };
  auto zerof = [&](float* p, size_t elems) {
    hipMemsetAsync(p, 0, elems * sizeof(float), stream);
  };
  auto bn = [&](const float* x, int n, int c, int kpad,
                const float* g, const float* b, unsigned short* ybf) {
    bn_stats<<<c, 256, 0, stream>>>(x, n, c, g, b, ss, ss + 256);
    bn_apply<<<gblocks((size_t)n * kpad), 256, 0, stream>>>(x, n, c, kpad, ss, ss + 256, ybf);
  };
  auto conv_wt = [&](const float* w, int KO, int cin, int cout, int kpad) -> unsigned short* {
    unsigned short* wt = (unsigned short*)alloc((size_t)KO * cout * kpad * 2);
    wconv_bf16t<<<gblocks((size_t)KO * cout * kpad), 256, 0, stream>>>(w, KO, cin, cout, kpad, wt);
    return wt;
  };
  auto gemm = [&](const unsigned short* xbf, const unsigned short* wtk,
                  const int* iin, const int* iout, int np, int kpad, int cout,
                  float* out, int ostride, int coff) {
    if (np <= 0) return;
    dim3 g((np + 15) / 16, (cout + 31) / 32);   // 2 col tiles per wave
    gather_gemm_scatter_wmma<<<g, 32, 0, stream>>>(xbf, wtk, iin, iout, np, kpad, cout,
                                                   out, ostride, coff);
  };

  // ---- initial conv: features(3) -> 16 over sm[0] ----
  feat2bf<<<gblocks((size_t)N0 * 32), 256, 0, stream>>>(features, N0, in0);
  {
    unsigned short* wt = conv_wt(w0, 27, 3, 16, 32);
    zerof(x0, (size_t)N0 * 16);
    for (int k = 0; k < 27; ++k)
      gemm(in0, wt + (size_t)k * 16 * 32, sm_in[0][k], sm_out[0][k], sm_n[0][k],
           32, 16, x0, 16, 0);
  }

  // ---- down pass ----
  const float* xin = x0;
  for (int l = 0; l < NLEV; ++l) {
    const int Cc = C[l], K = kp(Cc);
    bn(xin, Ns[l], Cc, K, lv[l].pre_g, lv[l].pre_b, buf[l]);
    unsigned short* wt = conv_wt(lv[l].pre_w, 27, Cc, Cc, K);
    zerof(xA[l], (size_t)Ns[l] * Cc);
    for (int k = 0; k < 27; ++k)
      gemm(buf[l], wt + (size_t)k * Cc * K, sm_in[l][k], sm_out[l][k], sm_n[l][k],
           K, Cc, xA[l], Cc, 0);
    if (l < NLEV - 1) {
      const int C2 = C[l + 1];
      bn(xA[l], Ns[l], Cc, K, lv[l].down_g, lv[l].down_b, buf[l]);
      unsigned short* wtd = conv_wt(lv[l].down_w, 8, Cc, C2, K);
      zerof(xd, (size_t)Ns[l + 1] * C2);
      for (int k = 0; k < 8; ++k)
        gemm(buf[l], wtd + (size_t)k * C2 * K, dn_idx[l][k], dn_par[l][k], dn_n[l][k],
             K, C2, xd, C2, 0);
      xin = xd;    // consumed by next level's pre-BN before xd is reused
    }
  }

  // ---- up pass ----
  const float* r = xA[6];
  for (int l = NLEV - 2; l >= 0; --l) {
    const int Cc = C[l], C2 = C[l + 1];
    const int K2 = kp(C2), Kc = kp(2 * Cc);
    bn(r, Ns[l + 1], C2, K2, lv[l].up_g, lv[l].up_b, buf[l + 1]);
    unsigned short* wtu = conv_wt(lv[l].up_w, 8, C2, Cc, K2);
    zerof(cat, (size_t)Ns[l] * 2 * Cc);
    copy_cols<<<gblocks((size_t)Ns[l] * Cc), 256, 0, stream>>>(xA[l], Ns[l], Cc,
                                                               cat, 2 * Cc, 0);
    for (int k = 0; k < 8; ++k)        // deconv: parent -> fine, into cat[:, Cc:2Cc]
      gemm(buf[l + 1], wtu + (size_t)k * Cc * K2, dn_par[l][k], dn_idx[l][k], dn_n[l][k],
           K2, Cc, cat, 2 * Cc, Cc);
    bn(cat, Ns[l], 2 * Cc, Kc, lv[l].post_g, lv[l].post_b, buf[l]);
    unsigned short* wtp = conv_wt(lv[l].post_w, 27, 2 * Cc, Cc, Kc);
    zerof(xP, (size_t)Ns[l] * Cc);
    for (int k = 0; k < 27; ++k)
      gemm(buf[l], wtp + (size_t)k * Cc * Kc, sm_in[l][k], sm_out[l][k], sm_n[l][k],
           Kc, Cc, xP, Cc, 0);
    r = xP;       // fully consumed by next iteration's up-BN before rewrite
  }

  // ---- final BN + linear ----
  bn(r, N0, 16, 32, fin_g, fin_b, buf[0]);
  linear_out<<<(N0 + 255) / 256, 256, 0, stream>>>(buf[0], N0, lin_w, lin_b,
                                                   (float*)d_out);
}
